// MemTransformerLM_61856118997491
// MI455X (gfx1250) — matrix operand order repaired and checked by reading, not compiled, not run on hardware
//
#include <hip/hip_runtime.h>
#include <math.h>

// ---------------------------------------------------------------------------
// MemTransformerLM forward on gfx1250 (MI455X).
// All large GEMMs run on v_wmma_f32_16x16x32_bf16 (bf16 in, fp32 accumulate).
// Roofline: ~660 GFLOP total vs ~280MB weights => compute-bound; WMMA path.
// ---------------------------------------------------------------------------

#define S_LEN   1024
#define BSZ     4
#define DMODEL  1024
#define NHEAD   16
#define DHEAD   64
#define DINNER  4096
#define NTOK    16000
#define NLAYER  4
#define HDH     (NHEAD * DHEAD)   // 1024
#define QKV3    (3 * HDH)         // 3072
#define ROWS    (S_LEN * BSZ)     // 4096

typedef __bf16 bf16;
typedef __attribute__((ext_vector_type(16))) __bf16 v16bf;
typedef __attribute__((ext_vector_type(8)))  __bf16 v8bf;
typedef __attribute__((ext_vector_type(8)))  float  v8f;

__device__ __forceinline__ bf16 bf_zero() {
  unsigned short z = 0;
  return __builtin_bit_cast(bf16, z);
}

// round-to-nearest-even fp32 -> bf16
__device__ __forceinline__ bf16 f2bf(float f) {
  unsigned u = __builtin_bit_cast(unsigned, f);
  unsigned r = (u + 0x7fffu + ((u >> 16) & 1u)) >> 16;
  unsigned short h = (unsigned short)r;
  return __builtin_bit_cast(bf16, h);
}

__device__ __forceinline__ v8bf v8bf_zero() {
  v8bf r;
#pragma unroll
  for (int i = 0; i < 8; ++i) r[i] = bf_zero();
  return r;
}

// ---------------------------------------------------------------------------
// Generic batched WMMA GEMM:  C[z] = act(A[z](MxK,bf16) * B[z](KxN,bf16) + bias)
// Block tile 128x128x32, 256 threads = 8 waves (4 m-waves x 2 n-waves),
// each wave computes 32x64 via 2x4 v_wmma_f32_16x16x32_bf16 accumulators.
// Requirements used by all call sites: K % 32 == 0, N % 8 == 0,
// 16B-aligned base pointers, row-major operands.
// ---------------------------------------------------------------------------
#define TM 128
#define TN 128
#define TKB 32
#define LDP 40   // padded LDS row stride in bf16 (80B, multiple of 16B)

__global__ void __launch_bounds__(256)
gemm_bf16_kernel(const bf16* __restrict__ Ag, long long strideA, int lda,
                 const bf16* __restrict__ Bg, long long strideB, int ldb, int bMod,
                 float* __restrict__ Cg, bf16* __restrict__ Cbg,
                 long long strideC, int ldc,
                 const float* __restrict__ bias,
                 int M, int N, int K, int act)
{
  __shared__ bf16 sA[TM * LDP];
  __shared__ bf16 sB[TN * LDP];

  const int tid     = threadIdx.x;
  const int lane    = tid & 31;
  const int wave    = tid >> 5;
  const int wm      = wave & 3;   // 0..3  -> 32-row slice
  const int wn      = wave >> 2;  // 0..1  -> 64-col slice
  const int laneLow = lane & 15;
  const int laneHi  = lane >> 4;

  const long long z = blockIdx.z;
  const bf16* A = Ag + z * strideA;
  const bf16* B = Bg + (bMod ? (long long)(z % bMod) : z) * strideB;

  const int mBase = blockIdx.y * TM;
  const int nBase = blockIdx.x * TN;

  v8f acc[2][4];
#pragma unroll
  for (int mi = 0; mi < 2; ++mi)
#pragma unroll
    for (int ni = 0; ni < 4; ++ni)
#pragma unroll
      for (int e = 0; e < 8; ++e) acc[mi][ni][e] = 0.f;

  for (int k0 = 0; k0 < K; k0 += TKB) {
    // ---- stage A tile (128 x 32) row-major into LDS ----
#pragma unroll
    for (int it = 0; it < 2; ++it) {
      int c   = tid + it * 256;      // 0..511 chunks of 8 bf16
      int row = c >> 2;              // 0..127
      int kc  = (c & 3) << 3;        // 0,8,16,24
      v8bf av;
      if (mBase + row < M) {
        const bf16* p = A + (long long)(mBase + row) * lda + k0 + kc;
        av = *(const v8bf*)p;
        if (k0 + TKB < K) __builtin_prefetch(p + TKB, 0, 1);  // global_prefetch_b8
      } else {
        av = v8bf_zero();
      }
      *(v8bf*)(&sA[row * LDP + kc]) = av;
    }
    // ---- stage B tile (32 x 128) transposed into LDS (Bt: 128 x 32) ----
#pragma unroll
    for (int it = 0; it < 2; ++it) {
      int c  = tid + it * 256;
      int kr = c >> 4;               // 0..31
      int nc = (c & 15) << 3;        // 0..120
      v8bf bv;
      if (nBase + nc < N) {
        const bf16* p = B + (long long)(k0 + kr) * ldb + nBase + nc;
        bv = *(const v8bf*)p;
        if (k0 + TKB < K) __builtin_prefetch(p + (long long)TKB * ldb, 0, 1);
      } else {
        bv = v8bf_zero();
      }
#pragma unroll
      for (int e = 0; e < 8; ++e) sB[(nc + e) * LDP + kr] = bv[e];
    }
    __syncthreads();

    // ---- operand fragments ----
    // A (16x32 bf16): lane m in 0..15 holds K = laneHi*8 + {0..7, 16..23}
    v16bf afr[2];
#pragma unroll
    for (int mi = 0; mi < 2; ++mi) {
      const bf16* p = &sA[(wm * 32 + mi * 16 + laneLow) * LDP + laneHi * 8];
      v8bf lo = *(const v8bf*)p;
      v8bf hi = *(const v8bf*)(p + 16);
#pragma unroll
      for (int e = 0; e < 8; ++e) { afr[mi][e] = lo[e]; afr[mi][8 + e] = hi[e]; }
    }
    // B (32x16 bf16): lane n holds K = laneHi*8 + {0..7, 16..23} of column n
    v16bf bfr[4];
#pragma unroll
    for (int ni = 0; ni < 4; ++ni) {
      const bf16* p = &sB[(wn * 64 + ni * 16 + laneLow) * LDP + laneHi * 8];
      v8bf lo = *(const v8bf*)p;
      v8bf hi = *(const v8bf*)(p + 16);
#pragma unroll
      for (int e = 0; e < 8; ++e) { bfr[ni][e] = lo[e]; bfr[ni][8 + e] = hi[e]; }
    }

#pragma unroll
    for (int mi = 0; mi < 2; ++mi)
#pragma unroll
      for (int ni = 0; ni < 4; ++ni)
        acc[mi][ni] = __builtin_amdgcn_wmma_f32_16x16x32_bf16(
            false, afr[mi], false, bfr[ni], (short)0, acc[mi][ni], false, false);
    __syncthreads();
  }

  // ---- epilogue: C/D layout: row = vr + 8*laneHi, col = laneLow ----
  float* C  = Cg  ? Cg  + z * strideC : nullptr;
  bf16*  Cb = Cbg ? Cbg + z * strideC : nullptr;
#pragma unroll
  for (int mi = 0; mi < 2; ++mi)
#pragma unroll
    for (int ni = 0; ni < 4; ++ni) {
      int gn = nBase + wn * 64 + ni * 16 + laneLow;
      if (gn >= N) continue;
      float bv = bias ? bias[gn] : 0.f;
#pragma unroll
      for (int vr = 0; vr < 8; ++vr) {
        int gm = mBase + wm * 32 + mi * 16 + laneHi * 8 + vr;
        if (gm >= M) continue;
        float v = acc[mi][ni][vr] + bv;
        if (act == 1) v = 0.5f * v * (1.f + erff(v * 0.70710678118654752f)); // exact GELU
        long long off = (long long)gm * ldc + gn;
        if (C)  C[off]  = v;
        if (Cb) Cb[off] = f2bf(v);
      }
    }
}

// ---------------------------------------------------------------------------
// Elementwise / support kernels
// ---------------------------------------------------------------------------
__global__ void k_embed(const int* __restrict__ tok, const float* __restrict__ emb,
                        float* __restrict__ h, bf16* __restrict__ hb)
{
  long long idx = (long long)blockIdx.x * 256 + threadIdx.x;
  if (idx >= (long long)ROWS * DMODEL) return;
  int d  = (int)(idx & (DMODEL - 1));
  int rb = (int)(idx >> 10);                 // s*B + b
  int t  = tok[rb];
  float v = emb[(long long)t * DMODEL + d];
  h[idx]  = v;
  hb[idx] = f2bf(v);
}

__global__ void k_posenc(bf16* __restrict__ rb)
{
  long long idx = (long long)blockIdx.x * 256 + threadIdx.x;
  if (idx >= (long long)S_LEN * DMODEL) return;
  int d = (int)(idx & (DMODEL - 1));
  int s = (int)(idx >> 10);
  float pos = (float)(S_LEN - 1 - s);
  int half = DMODEL / 2;
  int k = (d < half) ? d : d - half;
  float invf = expf(-(2.0f * (float)k / (float)DMODEL) * 9.210340371976184f); // ln 1e4
  float ang = pos * invf;
  rb[idx] = f2bf((d < half) ? sinf(ang) : cosf(ang));
}

__global__ void k_f2bf(const float* __restrict__ in, bf16* __restrict__ out, long long n)
{
  long long idx = (long long)blockIdx.x * 256 + threadIdx.x;
  if (idx < n) out[idx] = f2bf(in[idx]);
}

// qkv [S,B,3*H*DH] fp32 -> q1=q+r_w_bias, q2=q+r_r_bias (both [b,h,s,dh]),
// kT [b,h,dh,s], v [b,h,s,dh], all bf16
__global__ void k_split_qkv(const float* __restrict__ qkv, const float* __restrict__ qb,
                            const float* __restrict__ rwb, const float* __restrict__ rrb,
                            bf16* __restrict__ q1, bf16* __restrict__ q2,
                            bf16* __restrict__ kT, bf16* __restrict__ vv)
{
  long long idx = (long long)blockIdx.x * 256 + threadIdx.x;
  if (idx >= (long long)S_LEN * BSZ * NHEAD * DHEAD) return;
  int dh = (int)(idx & 63);
  int h  = (int)((idx >> 6) & 15);
  int b  = (int)((idx >> 10) & 3);
  int s  = (int)(idx >> 12);
  long long base = (long long)(s * BSZ + b) * QKV3;
  int hd = h * DHEAD + dh;
  float q = qkv[base + hd]           + qb[hd];
  float k = qkv[base + HDH + hd]     + qb[HDH + hd];
  float v = qkv[base + 2 * HDH + hd] + qb[2 * HDH + hd];
  long long zsd = (long long)(b * NHEAD + h) * S_LEN * DHEAD;
  q1[zsd + (long long)s * DHEAD + dh] = f2bf(q + rwb[hd]);
  q2[zsd + (long long)s * DHEAD + dh] = f2bf(q + rrb[hd]);
  kT[zsd + (long long)dh * S_LEN + s] = f2bf(k);
  vv[zsd + (long long)s * DHEAD + dh] = f2bf(v);
}

// rk [S, H*DH] fp32 -> rkT [h][dh][S] bf16
__global__ void k_rkT(const float* __restrict__ rk, bf16* __restrict__ rkT)
{
  long long idx = (long long)blockIdx.x * 256 + threadIdx.x;
  if (idx >= (long long)S_LEN * HDH) return;
  int s = (int)(idx >> 10);
  int r = (int)(idx & (HDH - 1));   // h*DHEAD + dh
  rkT[(long long)r * S_LEN + s] = f2bf(rk[idx]);
}

// score[i,j] = scale*(AC[i,j] + BD[i, j-i+S-1]) for j<=i else -inf; softmax -> bf16
__global__ void __launch_bounds__(256)
k_softmax(const float* __restrict__ AC, const float* __restrict__ BD,
          bf16* __restrict__ prob, float scale)
{
  int i = blockIdx.x;
  long long z = blockIdx.y;
  long long rbase = (z * S_LEN + i) * (long long)S_LEN;
  const float* ac = AC + rbase;
  const float* bd = BD + rbase;
  bf16* pr = prob + rbase;

  float vals[4];
  float mx = -3.4e38f;
#pragma unroll
  for (int t = 0; t < 4; ++t) {
    int j = threadIdx.x + t * 256;
    float v = -3.4e38f;
    if (j <= i) v = scale * (ac[j] + bd[j - i + S_LEN - 1]);
    vals[t] = v;
    mx = fmaxf(mx, v);
  }
  __shared__ float red[256];
  red[threadIdx.x] = mx;
  __syncthreads();
  for (int s = 128; s > 0; s >>= 1) {
    if (threadIdx.x < s) red[threadIdx.x] = fmaxf(red[threadIdx.x], red[threadIdx.x + s]);
    __syncthreads();
  }
  mx = red[0];
  __syncthreads();
  float sum = 0.f;
#pragma unroll
  for (int t = 0; t < 4; ++t) {
    float e = (vals[t] > -1e37f) ? expf(vals[t] - mx) : 0.f;
    vals[t] = e;
    sum += e;
  }
  red[threadIdx.x] = sum;
  __syncthreads();
  for (int s = 128; s > 0; s >>= 1) {
    if (threadIdx.x < s) red[threadIdx.x] += red[threadIdx.x + s];
    __syncthreads();
  }
  float inv = 1.f / red[0];
#pragma unroll
  for (int t = 0; t < 4; ++t) {
    int j = threadIdx.x + t * 256;
    pr[j] = f2bf(vals[t] * inv);   // masked lanes: vals==0
  }
}

// vec [b,h,s,dh] fp32 -> attA [(s*B+b), h*DH+dh] bf16
__global__ void k_vec_reorder(const float* __restrict__ vec, bf16* __restrict__ attA)
{
  long long idx = (long long)blockIdx.x * 256 + threadIdx.x;
  if (idx >= (long long)BSZ * NHEAD * S_LEN * DHEAD) return;
  int dh = (int)(idx & 63);
  int s  = (int)((idx >> 6) & 1023);
  int h  = (int)((idx >> 16) & 15);
  int b  = (int)(idx >> 20);
  attA[(long long)(s * BSZ + b) * HDH + h * DHEAD + dh] = f2bf(vec[idx]);
}

// h = LayerNorm(h + x) * g + b ; also writes bf16 copy
__global__ void __launch_bounds__(256)
k_add_ln(float* __restrict__ h, const float* __restrict__ x,
         const float* __restrict__ g, const float* __restrict__ b,
         bf16* __restrict__ hb)
{
  long long row = blockIdx.x;
  float* hr = h + row * DMODEL;
  const float* xr = x + row * DMODEL;

  float vals[4];
  float sum = 0.f;
#pragma unroll
  for (int t = 0; t < 4; ++t) {
    int j = threadIdx.x + t * 256;
    float v = hr[j] + xr[j];
    vals[t] = v;
    sum += v;
  }
  __shared__ float red[256];
  red[threadIdx.x] = sum;
  __syncthreads();
  for (int s = 128; s > 0; s >>= 1) {
    if (threadIdx.x < s) red[threadIdx.x] += red[threadIdx.x + s];
    __syncthreads();
  }
  float mu = red[0] * (1.f / DMODEL);
  __syncthreads();
  float vs = 0.f;
#pragma unroll
  for (int t = 0; t < 4; ++t) {
    float d = vals[t] - mu;
    vs += d * d;
  }
  red[threadIdx.x] = vs;
  __syncthreads();
  for (int s = 128; s > 0; s >>= 1) {
    if (threadIdx.x < s) red[threadIdx.x] += red[threadIdx.x + s];
    __syncthreads();
  }
  float inv = rsqrtf(red[0] * (1.f / DMODEL) + 1e-5f);
#pragma unroll
  for (int t = 0; t < 4; ++t) {
    int j = threadIdx.x + t * 256;
    float o = (vals[t] - mu) * inv * g[j] + b[j];
    hr[j] = o;
    hb[row * DMODEL + j] = f2bf(o);
  }
}

// ---------------------------------------------------------------------------
// Host side
// ---------------------------------------------------------------------------
static inline void gemm(hipStream_t st,
                        const bf16* A, long long sA, int lda,
                        const bf16* B, long long sB, int ldb, int bMod,
                        float* C, bf16* Cb, long long sC, int ldc,
                        const float* bias, int M, int N, int K, int act, int batch)
{
  dim3 grid((N + TN - 1) / TN, (M + TM - 1) / TM, batch);
  gemm_bf16_kernel<<<grid, 256, 0, st>>>(A, sA, lda, B, sB, ldb, bMod,
                                         C, Cb, sC, ldc, bias, M, N, K, act);
}

static inline long long blks(long long n) { return (n + 255) / 256; }

extern "C" void kernel_launch(void* const* d_in, const int* in_sizes, int n_in,
                              void* d_out, int out_size, void* d_ws, size_t ws_size,
                              hipStream_t stream)
{
  const int*   tokens   = (const int*)  d_in[0];
  const float* word_emb = (const float*)d_in[1];
  const float* r_w_bias = (const float*)d_in[2];
  const float* r_r_bias = (const float*)d_in[3];
  const float* qkv_w    = (const float*)d_in[4];
  const float* qkv_b    = (const float*)d_in[5];
  const float* r_kw     = (const float*)d_in[6];
  const float* r_kb     = (const float*)d_in[7];
  const float* o_w      = (const float*)d_in[8];
  const float* o_b      = (const float*)d_in[9];
  const float* ln1_g    = (const float*)d_in[10];
  const float* ln1_b    = (const float*)d_in[11];
  const float* ff1_w    = (const float*)d_in[12];
  const float* ff1_b    = (const float*)d_in[13];
  const float* ff2_w    = (const float*)d_in[14];
  const float* ff2_b    = (const float*)d_in[15];
  const float* ln2_g    = (const float*)d_in[16];
  const float* ln2_b    = (const float*)d_in[17];
  const float* final_w  = (const float*)d_in[18];
  const float* final_b  = (const float*)d_in[19];
  float* out = (float*)d_out;

  // ---- workspace bump allocator (256B aligned) ----
  char* ws = (char*)d_ws;
  size_t off = 0;
  auto alloc = [&](long long elems, size_t elsz) -> void* {
    off = (off + 255) & ~(size_t)255;
    void* p = ws + off;
    off += (size_t)elems * elsz;
    return p;
  };

  bf16* wq  = (bf16*)alloc((long long)NLAYER * DMODEL * QKV3, 2);
  bf16* wr  = (bf16*)alloc((long long)NLAYER * DMODEL * HDH, 2);
  bf16* wo  = (bf16*)alloc((long long)NLAYER * HDH * DMODEL, 2);
  bf16* w1  = (bf16*)alloc((long long)NLAYER * DMODEL * DINNER, 2);
  bf16* w2  = (bf16*)alloc((long long)NLAYER * DINNER * DMODEL, 2);
  bf16* wf  = (bf16*)alloc((long long)DMODEL * NTOK, 2);

  float* hf   = (float*)alloc((long long)ROWS * DMODEL, 4);
  bf16*  hb   = (bf16*) alloc((long long)ROWS * DMODEL, 2);
  bf16*  rpe  = (bf16*) alloc((long long)S_LEN * DMODEL, 2);
  float* qkvf = (float*)alloc((long long)ROWS * QKV3, 4);
  bf16*  q1   = (bf16*) alloc((long long)BSZ * NHEAD * S_LEN * DHEAD, 2);
  bf16*  q2   = (bf16*) alloc((long long)BSZ * NHEAD * S_LEN * DHEAD, 2);
  bf16*  kT   = (bf16*) alloc((long long)BSZ * NHEAD * DHEAD * S_LEN, 2);
  bf16*  vb   = (bf16*) alloc((long long)BSZ * NHEAD * S_LEN * DHEAD, 2);
  float* rkf  = (float*)alloc((long long)S_LEN * HDH, 4);
  bf16*  rkT  = (bf16*) alloc((long long)HDH * S_LEN, 2);
  float* scAC = (float*)alloc((long long)BSZ * NHEAD * S_LEN * S_LEN, 4);
  float* scBD = (float*)alloc((long long)BSZ * NHEAD * S_LEN * S_LEN, 4);
  bf16*  prob = (bf16*) alloc((long long)BSZ * NHEAD * S_LEN * S_LEN, 2);
  float* vecf = (float*)alloc((long long)BSZ * NHEAD * S_LEN * DHEAD, 4);
  bf16*  attA = (bf16*) alloc((long long)ROWS * HDH, 2);
  float* attn = (float*)alloc((long long)ROWS * DMODEL, 4);
  bf16*  ff1o = (bf16*) alloc((long long)ROWS * DINNER, 2);
  float* ff2o = (float*)alloc((long long)ROWS * DMODEL, 4);
  (void)ws_size; (void)in_sizes; (void)n_in; (void)out_size;

  const long long NB = BSZ * NHEAD;  // 64 attention batches

  // ---- weight conversion fp32 -> bf16 (deterministic every call) ----
  {
    long long n;
    n = (long long)NLAYER * DMODEL * QKV3;  k_f2bf<<<blks(n), 256, 0, stream>>>(qkv_w,  wq, n);
    n = (long long)NLAYER * DMODEL * HDH;   k_f2bf<<<blks(n), 256, 0, stream>>>(r_kw,   wr, n);
    n = (long long)NLAYER * HDH * DMODEL;   k_f2bf<<<blks(n), 256, 0, stream>>>(o_w,    wo, n);
    n = (long long)NLAYER * DMODEL * DINNER;k_f2bf<<<blks(n), 256, 0, stream>>>(ff1_w,  w1, n);
    n = (long long)NLAYER * DINNER * DMODEL;k_f2bf<<<blks(n), 256, 0, stream>>>(ff2_w,  w2, n);
    n = (long long)DMODEL * NTOK;           k_f2bf<<<blks(n), 256, 0, stream>>>(final_w,wf, n);
  }

  // ---- embedding + positional encoding ----
  k_embed<<<blks((long long)ROWS * DMODEL), 256, 0, stream>>>(tokens, word_emb, hf, hb);
  k_posenc<<<blks((long long)S_LEN * DMODEL), 256, 0, stream>>>(rpe);

  for (int l = 0; l < NLAYER; ++l) {
    // qkv = h @ qkv_w[l] + qkv_b[l]
    gemm(stream, hb, 0, DMODEL, wq + (long long)l * DMODEL * QKV3, 0, QKV3, 0,
         qkvf, nullptr, 0, QKV3, qkv_b + (long long)l * QKV3,
         ROWS, QKV3, DMODEL, 0, 1);
    k_split_qkv<<<blks((long long)S_LEN * BSZ * NHEAD * DHEAD), 256, 0, stream>>>(
        qkvf, qkv_b + (long long)l * QKV3, r_w_bias, r_r_bias, q1, q2, kT, vb);

    // rk = r @ r_kw[l] + r_kb[l];  then transpose per-head
    gemm(stream, rpe, 0, DMODEL, wr + (long long)l * DMODEL * HDH, 0, HDH, 0,
         rkf, nullptr, 0, HDH, r_kb + (long long)l * HDH,
         S_LEN, HDH, DMODEL, 0, 1);
    k_rkT<<<blks((long long)S_LEN * HDH), 256, 0, stream>>>(rkf, rkT);

    // AC[z] = q1[z] @ kT[z]   (z = b*H + h), M=N=S, K=DH
    gemm(stream, q1, (long long)S_LEN * DHEAD, DHEAD,
         kT, (long long)DHEAD * S_LEN, S_LEN, 0,
         scAC, nullptr, (long long)S_LEN * S_LEN, S_LEN, nullptr,
         S_LEN, S_LEN, DHEAD, 0, (int)NB);
    // BD[z] = q2[z] @ rkT[h]  (B batch indexed by z % H)
    gemm(stream, q2, (long long)S_LEN * DHEAD, DHEAD,
         rkT, (long long)DHEAD * S_LEN, S_LEN, NHEAD,
         scBD, nullptr, (long long)S_LEN * S_LEN, S_LEN, nullptr,
         S_LEN, S_LEN, DHEAD, 0, (int)NB);

    // masked softmax with fused rel_shift; scale = 1/sqrt(DH)
    {
      dim3 g(S_LEN, (unsigned)NB);
      k_softmax<<<g, 256, 0, stream>>>(scAC, scBD, prob, 0.125f);
    }

    // vec[z] = prob[z] @ v[z]  (M=S, N=DH, K=S)
    gemm(stream, prob, (long long)S_LEN * S_LEN, S_LEN,
         vb, (long long)S_LEN * DHEAD, DHEAD, 0,
         vecf, nullptr, (long long)S_LEN * DHEAD, DHEAD, nullptr,
         S_LEN, DHEAD, S_LEN, 0, (int)NB);
    k_vec_reorder<<<blks((long long)NB * S_LEN * DHEAD), 256, 0, stream>>>(vecf, attA);

    // attn_out = vec @ o_w[l] + o_b[l]
    gemm(stream, attA, 0, HDH, wo + (long long)l * HDH * DMODEL, 0, DMODEL, 0,
         attn, nullptr, 0, DMODEL, o_b + (long long)l * DMODEL,
         ROWS, DMODEL, HDH, 0, 1);

    // h = LN(h + attn_out)
    k_add_ln<<<ROWS, 256, 0, stream>>>(hf, attn,
        ln1_g + (long long)l * DMODEL, ln1_b + (long long)l * DMODEL, hb);

    // ff1 = gelu(h @ ff1_w[l] + ff1_b[l])  (bf16 out only)
    gemm(stream, hb, 0, DMODEL, w1 + (long long)l * DMODEL * DINNER, 0, DINNER, 0,
         nullptr, ff1o, 0, DINNER, ff1_b + (long long)l * DINNER,
         ROWS, DINNER, DMODEL, 1, 1);
    // ff2 = ff1 @ ff2_w[l] + ff2_b[l]
    gemm(stream, ff1o, 0, DINNER, w2 + (long long)l * DINNER * DMODEL, 0, DMODEL, 0,
         ff2o, nullptr, 0, DMODEL, ff2_b + (long long)l * DMODEL,
         ROWS, DMODEL, DINNER, 0, 1);

    // h = LN(h + ff2)
    k_add_ln<<<ROWS, 256, 0, stream>>>(hf, ff2o,
        ln2_g + (long long)l * DMODEL, ln2_b + (long long)l * DMODEL, hb);
  }

  // logits = h @ final_w + final_b  -> d_out [S,B,V]
  gemm(stream, hb, 0, DMODEL, wf, 0, NTOK, 0,
       out, nullptr, 0, NTOK, final_b,
       ROWS, NTOK, DMODEL, 0, 1);
}
